// SModel_81552839016873
// MI455X (gfx1250) — matrix operand order repaired and checked by reading, not compiled, not run on hardware
//
#include <hip/hip_runtime.h>
#include <hip/hip_bf16.h>

// ---------------------------------------------------------------------------
// GRU (B=512, T=512, I=8, H=256) for gfx1250 (MI455X).
// One persistent workgroup per 16 batch rows (32 WGs); the T=512 recurrence
// runs inside the kernel with h kept in LDS (fp32 state + bf16 WMMA copy).
// Recurrent+input GEMM uses v_wmma_f32_16x16x32_bf16.
// Weight fragments: 40/54 held persistently in VGPRs (explicit budget, no
// spills), 14/54 streamed per step from the L2-resident packed buffer through
// a laundered addrspace(1) pointer: LICM cannot hoist them (no spills) and
// they issue as global_load_b128 (LOADcnt only, not the FLAT LOADcnt+DScnt
// path that would couple them into the A-fragment ds waits).
// Pipeline per step: prefetch x_{t+1} -> WMMA accum -> barrier ->
// gate update + stage x_{t+1} -> barrier -> prediction.
// ---------------------------------------------------------------------------

#define Bsz 512
#define Tst 512
#define Isz 8
#define Hsz 256
#define H3  768
#define NCHUNK 9            // K = 288 = 256 (h) + 32 (x padded from 8)
#define APITCH 296          // bf16 elems; 148 dwords % 64 == 20 -> conflict-free b128
#define HPITCH 260          // fp32 elems

typedef __attribute__((ext_vector_type(16))) __bf16        bf16x16;
typedef __attribute__((ext_vector_type(8)))  float         f32x8;
typedef __attribute__((ext_vector_type(4)))  unsigned int  u32x4;

typedef const __attribute__((address_space(1))) unsigned short gus_t;
typedef const __attribute__((address_space(1))) u32x4          gu32x4_t;

__device__ __forceinline__ unsigned short f2bf(float f) {
    unsigned int u = __float_as_uint(f);
    u += 0x7FFFu + ((u >> 16) & 1u);            // round-to-nearest-even
    return (unsigned short)(u >> 16);
}

__device__ __forceinline__ float sigmoid_f(float x) {
    return 1.0f / (1.0f + __expf(-x));
}
__device__ __forceinline__ float tanh_f(float x) {
    x = fminf(fmaxf(x, -15.0f), 15.0f);
    float e = __expf(-2.0f * x);
    return (1.0f - e) / (1.0f + e);
}

// Assemble a 32-byte WMMA fragment (16 bf16) from two 16B chunks (generic ptr).
__device__ __forceinline__ bf16x16 ld_frag(const unsigned short* p) {
    struct { u32x4 lo, hi; } t;
    t.lo = *(const u32x4*)(p);
    t.hi = *(const u32x4*)(p + 8);
    bf16x16 v;
    __builtin_memcpy(&v, &t, 32);
    return v;
}

// Same, but through an explicit global (addrspace 1) pointer -> global_load_b128.
__device__ __forceinline__ bf16x16 ld_frag_g(gus_t* p) {
    struct { u32x4 lo, hi; } t;
    t.lo = *(gu32x4_t*)(p);
    t.hi = *(gu32x4_t*)(p + 8);
    bf16x16 v;
    __builtin_memcpy(&v, &t, 32);
    return v;
}

__device__ __forceinline__ f32x8 wmma_bf16(bf16x16 a, bf16x16 b, f32x8 c) {
    return __builtin_amdgcn_wmma_f32_16x16x32_bf16(
        false, a, false, b, (short)0, c, false, false);
}

// ---------------------------------------------------------------------------
// Setup: pack [W_hh^T ; W_ih^T ; zeros] (K=288 x N=768) as bf16 in contiguous-k (k = 16h + j)
// B-fragment order: wpack[((tile*9 + kc)*32 + lane)*16 + j].
// lane<16 -> n = tile*16+lane, k-offsets {0..15};
// lane>=16 -> same n, k-offsets {16..31} (transpose-dual of A layout).
// ---------------------------------------------------------------------------
__global__ void gru_pack_weights(const float* __restrict__ Wih,
                                 const float* __restrict__ Whh,
                                 unsigned short* __restrict__ wpack) {
    int idx = blockIdx.x * blockDim.x + threadIdx.x;   // 48*9*32*16 = 221184
    if (idx >= 48 * NCHUNK * 512) return;
    int j    =  idx        & 15;
    int lane = (idx >> 4)  & 31;
    int kc   = (idx >> 9)  % NCHUNK;
    int tile =  idx / (NCHUNK * 512);
    int n    = tile * 16 + (lane & 15);
    int koff = j + ((lane & 16) ? 16 : 0);
    int k    = kc * 32 + koff;
    float v = 0.0f;
    if (k < Hsz)            v = Whh[n * Hsz + k];
    else if (k < Hsz + Isz) v = Wih[n * Isz + (k - Hsz)];
    wpack[idx] = f2bf(v);
}

// ---------------------------------------------------------------------------
// Persistent GRU kernel: 32 blocks x 256 threads (8 wave32 waves).
// Wave w owns H-column tiles 2w and 2w+1 (32 of the 256 hidden columns).
// ---------------------------------------------------------------------------
__global__ void __launch_bounds__(256, 1)
gru_persistent(const float* __restrict__ x,
               const float* __restrict__ bih,
               const float* __restrict__ bhh,
               const float* __restrict__ Wlin,
               const float* __restrict__ blin,
               const unsigned short* __restrict__ wpack,
               float* __restrict__ out) {
    __shared__ __align__(16) unsigned short Abuf[16][APITCH]; // bf16 [h | x_t]
    __shared__ float h32[16][HPITCH];                         // fp32 state
    __shared__ float sbih[H3];
    __shared__ float sbhh[H3];
    __shared__ float swl[Hsz];

    const int tid  = threadIdx.x;
    const int wave = tid >> 5;
    const int lane = tid & 31;
    const int hi   = lane >> 4;       // half-wave select
    const int nl   = lane & 15;       // M (A-rows) / N (B-cols) lane index
    const int wg   = blockIdx.x;      // batch tile: rows wg*16 .. wg*16+15

    // ---- init LDS ----
    for (int i = tid; i < 16 * APITCH; i += 256) (&Abuf[0][0])[i] = 0;
    for (int i = tid; i < 16 * HPITCH; i += 256) (&h32[0][0])[i] = 0.0f;
    for (int i = tid; i < H3; i += 256) { sbih[i] = bih[i]; sbhh[i] = bhh[i]; }
    if (tid < Hsz) swl[tid] = Wlin[tid];
    const float bl = blin[0];

    // x-staging role: threads 0..127 each own one (row, input-col) slot.
    const int xm = tid >> 3, xi = tid & 7;
    const size_t xbase = ((size_t)(wg * 16 + xm) * Tst) * Isz + xi;

    // stage x_0
    if (tid < 128) Abuf[xm][Hsz + xi] = f2bf(x[xbase]);

    // ---- persistent register-resident B fragments (40 frags = 320 VGPRs) ----
    // r-gate k-chunks 0..7, z-gate 0..7, n-gate (h part) 0..3, for both tiles.
    const int ht0 = 2 * wave;
    bf16x16 Br[2][8], Bz[2][8], Bn[2][4];
    #pragma unroll
    for (int jj = 0; jj < 2; ++jj) {
        const unsigned short* pr = wpack + (size_t)(ht0 + jj)      * NCHUNK * 512 + lane * 16;
        const unsigned short* pz = wpack + (size_t)(16 + ht0 + jj) * NCHUNK * 512 + lane * 16;
        const unsigned short* pn = wpack + (size_t)(32 + ht0 + jj) * NCHUNK * 512 + lane * 16;
        #pragma unroll
        for (int kc = 0; kc < 8; ++kc) {
            Br[jj][kc] = ld_frag(pr + kc * 512);
            Bz[jj][kc] = ld_frag(pz + kc * 512);
        }
        #pragma unroll
        for (int kc = 0; kc < 4; ++kc)
            Bn[jj][kc] = ld_frag(pn + kc * 512);
    }
    __syncthreads();

    for (int t = 0; t < Tst; ++t) {
        // Launder the weight base address so the 14 streamed fragments cannot
        // be LICM-hoisted (that caused scratch spills), then recover an
        // explicit addrspace(1) pointer so they issue as global_load_b128.
        unsigned long long wpo = (unsigned long long)wpack;
        asm volatile("" : "+s"(wpo));
        gus_t* wps = (gus_t*)wpo;

        // ---- prefetch x_{t+1} into a register; completes under WMMA chain ----
        float xnext = 0.0f;
        const bool havex = (tid < 128) && (t + 1 < Tst);
        if (havex) xnext = x[xbase + (size_t)(t + 1) * Isz];

        const f32x8 vz = {0, 0, 0, 0, 0, 0, 0, 0};
        f32x8 acc_r[2]  = {vz, vz};
        f32x8 acc_z[2]  = {vz, vz};
        f32x8 acc_hn[2] = {vz, vz};
        f32x8 acc_in[2] = {vz, vz};

        // ---- h-part accumulation: A fragment loaded once per k-chunk,
        //      feeds 6 independent WMMA chains (r, z, hn x 2 tiles) ----
        #pragma unroll
        for (int kc = 0; kc < 8; ++kc) {
            const bf16x16 a = ld_frag(&Abuf[nl][kc * 32 + hi * 16]);
            #pragma unroll
            for (int jj = 0; jj < 2; ++jj) {
                acc_r[jj] = wmma_bf16(a, Br[jj][kc], acc_r[jj]);
                acc_z[jj] = wmma_bf16(a, Bz[jj][kc], acc_z[jj]);
                if (kc < 4) {
                    acc_hn[jj] = wmma_bf16(a, Bn[jj][kc], acc_hn[jj]);
                } else {    // streamed hn chunks 4..7
                    acc_hn[jj] = wmma_bf16(
                        a,
                        ld_frag_g(wps + ((size_t)(32 + ht0 + jj) * NCHUNK + kc) * 512 + lane * 16),
                        acc_hn[jj]);
                }
            }
        }
        // ---- x-part (k-chunk 8): r, z fused tail + i_n (all streamed) ----
        {
            const bf16x16 a8 = ld_frag(&Abuf[nl][8 * 32 + hi * 16]);
            #pragma unroll
            for (int jj = 0; jj < 2; ++jj) {
                acc_r[jj]  = wmma_bf16(a8,
                    ld_frag_g(wps + ((size_t)(ht0 + jj)      * NCHUNK + 8) * 512 + lane * 16),
                    acc_r[jj]);
                acc_z[jj]  = wmma_bf16(a8,
                    ld_frag_g(wps + ((size_t)(16 + ht0 + jj) * NCHUNK + 8) * 512 + lane * 16),
                    acc_z[jj]);
                acc_in[jj] = wmma_bf16(a8,
                    ld_frag_g(wps + ((size_t)(32 + ht0 + jj) * NCHUNK + 8) * 512 + lane * 16),
                    acc_in[jj]);
            }
        }
        __syncthreads();   // B1: all A reads done before h columns are rewritten

        // ---- stage x_{t+1} (A cols 256..263, disjoint from h cols 0..255) ----
        if (havex) Abuf[xm][Hsz + xi] = f2bf(xnext);

        // ---- gates + state update (C/D layout: lane->(n=nl, m=r+hi*8)) ----
        #pragma unroll
        for (int jj = 0; jj < 2; ++jj) {
            const int colH = (ht0 + jj) * 16 + nl;
            const float br  = sbih[colH]       + sbhh[colH];
            const float bz  = sbih[Hsz + colH] + sbhh[Hsz + colH];
            const float bin_ = sbih[2 * Hsz + colH];
            const float bhn_ = sbhh[2 * Hsz + colH];
            #pragma unroll
            for (int r = 0; r < 8; ++r) {
                const int m = r + hi * 8;
                const float hold = h32[m][colH];
                const float rg = sigmoid_f(acc_r[jj][r] + br);
                const float zg = sigmoid_f(acc_z[jj][r] + bz);
                const float ng = tanh_f((acc_in[jj][r] + bin_) +
                                        rg * (acc_hn[jj][r] + bhn_));
                const float hn2 = (1.0f - zg) * ng + zg * hold;
                h32[m][colH]  = hn2;
                Abuf[m][colH] = f2bf(hn2);
            }
        }
        __syncthreads();   // B2: h32/Abuf fully updated

        // ---- pred_t = h_new @ W_lin^T + b_lin  (wave w -> rows 2w, 2w+1) ----
        // pred(t) reads h32; next step's writes to h32 happen after B1(t+1).
        #pragma unroll
        for (int rr = 0; rr < 2; ++rr) {
            const int m = 2 * wave + rr;
            float s = 0.0f;
            #pragma unroll
            for (int i = 0; i < 8; ++i) {
                const int c = lane + 32 * i;
                s += h32[m][c] * swl[c];
            }
            #pragma unroll
            for (int off = 16; off > 0; off >>= 1)
                s += __shfl_xor(s, off, 32);
            if (lane == 0)
                out[(size_t)(wg * 16 + m) * Tst + t] = s + bl;
        }
    }
}

// ---------------------------------------------------------------------------
extern "C" void kernel_launch(void* const* d_in, const int* in_sizes, int n_in,
                              void* d_out, int out_size, void* d_ws, size_t ws_size,
                              hipStream_t stream) {
    const float* x    = (const float*)d_in[0];   // [512,512,8]
    const float* Wih  = (const float*)d_in[1];   // [768,8]
    const float* Whh  = (const float*)d_in[2];   // [768,256]
    const float* bih  = (const float*)d_in[3];   // [768]
    const float* bhh  = (const float*)d_in[4];   // [768]
    const float* Wlin = (const float*)d_in[5];   // [256]
    const float* blin = (const float*)d_in[6];   // [1]
    float* out = (float*)d_out;                  // [512,512] (B,T,1)

    unsigned short* wpack = (unsigned short*)d_ws;   // 48*9*512*2 = 442368 B

    const int packElems = 48 * NCHUNK * 512;
    gru_pack_weights<<<(packElems + 255) / 256, 256, 0, stream>>>(Wih, Whh, wpack);
    gru_persistent<<<Bsz / 16, 256, 0, stream>>>(x, bih, bhh, Wlin, blin, wpack, out);
}
